// SelfAttentionBlock_79645873537360
// MI455X (gfx1250) — matrix operand order repaired and checked by reading, not compiled, not run on hardware
//
#include <hip/hip_runtime.h>
#include <stdint.h>

typedef __attribute__((ext_vector_type(16))) __bf16 v16bf;
typedef __attribute__((ext_vector_type(8)))  float  v8f;
typedef int v4i_vs __attribute__((vector_size(16)));   // matches builtin param

#define AS_GLOBAL __attribute__((address_space(1)))
#define AS_LDS    __attribute__((address_space(3)))

static constexpr int kB = 4, kC = 64, kH = 512, kWd = 512;
static constexpr int kHW = kH * kWd;       // 262144
static constexpr int kHeads = kB * kC;     // 256

// padded LDS leading dims (bf16 elems): rows stay 16B-aligned (b128 merge),
// 1040B row stride = 4-bank rotation (vs 0 when unpadded)
static constexpr int LDQ = 520;   // Q tile 64x512
static constexpr int LDK = 520;   // K tile 32x512
static constexpr int LDV = 40;    // V^T tile 512x32 (80B rows, 16B aligned)

#if __has_builtin(__builtin_amdgcn_global_load_async_to_lds_b128)
#define ATTN_ASYNC 1
#endif

__device__ __forceinline__ void wait_async() {
#if __has_builtin(__builtin_amdgcn_s_wait_asynccnt)
  __builtin_amdgcn_s_wait_asynccnt(0);
#else
  asm volatile("s_wait_asynccnt 0" ::: "memory");
#endif
}

__device__ __forceinline__ unsigned short f32_bf16(float f) {
  union { float f; unsigned u; } x; x.f = f;
  unsigned r = x.u + 0x7FFFu + ((x.u >> 16) & 1u);   // round-nearest-even
  return (unsigned short)(r >> 16);
}

union FragB { v16bf v; unsigned short u[16]; unsigned d[8]; };
union FragC { v8f v; float f[8]; };

__device__ __forceinline__ v8f wmma_bf16(v16bf a, v16bf b, v8f c) {
  return __builtin_amdgcn_wmma_f32_16x16x32_bf16(false, a, false, b, (short)0, c,
                                                 false, false);
}

// A operand (16x32) from row-major [M][ld] bf16 in LDS.
// ISA layout: lanes0-15: V0-3 K=0..7, V4-7 K=16..23 ; lanes16-31: +8.
__device__ __forceinline__ v16bf a_frag(const unsigned short* p, int row0, int k0, int ld) {
  const int lane = threadIdx.x & 31, hf = lane >> 4;
  const unsigned short* base = p + (row0 + (lane & 15)) * ld + k0 + hf * 8;
  FragB f;
#pragma unroll
  for (int j = 0; j < 4; ++j) {
    f.d[j]     = *(const unsigned*)(base + 2 * j);
    f.d[4 + j] = *(const unsigned*)(base + 16 + 2 * j);
  }
  return f.v;
}

// B operand (32x16) where memory is n-major [N][ld], reduction dim contiguous:
// element(k,n) = p[n*ld + k]. lanes0-15: V0-3 K=0..7, V4-7 K=16..23 ; lanes16-31: +8.
__device__ __forceinline__ v16bf b_frag_nk(const unsigned short* p, int n0, int k0, int ld) {
  const int lane = threadIdx.x & 31, hf = lane >> 4;
  const unsigned short* base = p + (n0 + (lane & 15)) * ld + k0 + hf * 8;
  FragB f;
#pragma unroll
  for (int j = 0; j < 4; ++j) {
    f.d[j]     = *(const unsigned*)(base + 2 * j);
    f.d[4 + j] = *(const unsigned*)(base + 16 + 2 * j);
  }
  return f.v;
}

// Copy NROWSx512 bf16 tile global->LDS (row stride LDDST elems).
// Uses gfx1250 async-to-LDS (ASYNCcnt) when available; sync uint4 otherwise.
template <int NROWS, int LDDST>
__device__ __forceinline__ void tile_to_lds(const unsigned short* g,
                                            unsigned short* l, int tid) {
  constexpr int CH = NROWS * 64;          // 16-byte chunks
  for (int e = tid; e < CH; e += 256) {
    int row = e >> 6, c8 = (e & 63) * 8;
#ifdef ATTN_ASYNC
    __builtin_amdgcn_global_load_async_to_lds_b128(
        (AS_GLOBAL v4i_vs*)(g + (size_t)row * 512 + c8),
        (AS_LDS v4i_vs*)(l + row * LDDST + c8), 0, 0);
#else
    *(uint4*)(l + row * LDDST + c8) =
        *(const uint4*)(g + (size_t)row * 512 + c8);
#endif
  }
}

// ---------------------------------------------------------------- fold BN
__global__ void fold_kernel(const float* __restrict__ w, const float* __restrict__ b,
                            const float* __restrict__ gamma, const float* __restrict__ beta,
                            const float* __restrict__ mean, const float* __restrict__ var,
                            float extra, unsigned short* __restrict__ wo,
                            float* __restrict__ bo) {
  int i = blockIdx.x * blockDim.x + threadIdx.x;   // 0..4095
  int o = i >> 6, c = i & 63;
  float scale = gamma[o] * rsqrtf(var[o] + 1e-5f);
  wo[i] = f32_bf16(w[i] * scale * extra);
  if (c == 0) bo[o] = ((b[o] - mean[o]) * scale + beta[o]) * extra;
}

// ------------------------------------------------------- fused QKV conv+BN
__global__ void __launch_bounds__(256) qkv_kernel(
    const float* __restrict__ x,
    const unsigned short* __restrict__ wq, const unsigned short* __restrict__ wk,
    const unsigned short* __restrict__ wv,
    const float* __restrict__ bq, const float* __restrict__ bk,
    const float* __restrict__ bv,
    unsigned short* __restrict__ Q, unsigned short* __restrict__ K,
    unsigned short* __restrict__ V) {
  __shared__ unsigned short sW[3][64 * 64];
  __shared__ float sB[3][64];
  __shared__ unsigned short sX[128 * 72];   // [spatial][c], padded
  const int tid = threadIdx.x;
  const int bb = blockIdx.y;
  const int s0 = blockIdx.x * 128;

  for (int e = tid; e < 64 * 64; e += 256) {
    sW[0][e] = wq[e]; sW[1][e] = wk[e]; sW[2][e] = wv[e];
  }
  if (tid < 64) { sB[0][tid] = bq[tid]; sB[1][tid] = bk[tid]; sB[2][tid] = bv[tid]; }
  for (int e = tid; e < 64 * 128; e += 256) {
    int c = e >> 7, sl = e & 127;
    sX[sl * 72 + c] = f32_bf16(x[(size_t)(bb * 64 + c) * kHW + s0 + sl]);
  }
  __syncthreads();

  const int wave = tid >> 5, lane = tid & 31, hf = lane >> 4;
  const int nloc = wave * 16;
  const int spos = s0 + nloc + (lane & 15);
  unsigned short* const outp[3] = {Q, K, V};
#pragma unroll
  for (int set = 0; set < 3; ++set) {
#pragma unroll
    for (int ot = 0; ot < 4; ++ot) {
      const int ob = ot * 16;
      FragC acc;
#pragma unroll
      for (int r = 0; r < 8; ++r) acc.f[r] = sB[set][ob + r + 8 * hf];
#pragma unroll
      for (int kk = 0; kk < 2; ++kk) {
        v16bf a = a_frag(sW[set], ob, kk * 32, 64);
        v16bf bop = b_frag_nk(sX, nloc, kk * 32, 72);
        acc.v = wmma_bf16(a, bop, acc.v);
      }
#pragma unroll
      for (int r = 0; r < 8; ++r) {
        int o = ob + r + 8 * hf;
        outp[set][(size_t)(bb * 64 + o) * kHW + spos] = f32_bf16(acc.f[r]);
      }
    }
  }
}

// ------------------------------------------- flash attention per (head, q-tile)
// LDS: sQ 64x520 | sK[2] 32x520 | sVn[2] 32x512 | sVt 512x40 | sP 64x32 |
//      sS 64x32 f32 | sRow 64 f32  == 252,160 B (of 320 KB WGP LDS, 1 WG/WGP)
__global__ void __launch_bounds__(256) attn_kernel(
    const unsigned short* __restrict__ Q, const unsigned short* __restrict__ K,
    const unsigned short* __restrict__ V, unsigned short* __restrict__ O) {
  extern __shared__ __align__(16) char smem[];
  unsigned short* sQ  = (unsigned short*)smem;       // 64 x LDQ
  unsigned short* sKb = sQ + 64 * LDQ;               // 2 x 32 x LDK
  unsigned short* sVn = sKb + 2 * 32 * LDK;          // 2 x 32 x 512 (natural)
  unsigned short* sVt = sVn + 2 * 32 * 512;          // 512 x LDV (transposed)
  unsigned short* sP  = sVt + 512 * LDV;             // 64 x 32 bf16
  float* sS   = (float*)(sP + 64 * 32);              // 64 x 32 f32
  float* sRow = sS + 64 * 32;                        // 64 f32 (alpha / inv_l)

  const int tid = threadIdx.x;
  const int head = blockIdx.y;
  const int q0 = blockIdx.x * 64;
  const size_t hb = (size_t)head * kHW;

  // prologue: Q tile + first K/V tiles (async when available)
  tile_to_lds<64, LDQ>(Q + hb + (size_t)q0 * 512, sQ, tid);
  tile_to_lds<32, LDK>(K + hb, sKb, tid);
  tile_to_lds<32, 512>(V + hb, sVn, tid);

  const int wave = tid >> 5, lane = tid & 31, hf = lane >> 4;
  const int rg = (wave & 3) * 16;       // S/O row group
  const int cg16 = (wave >> 2) * 16;    // S col group
  const int cg256 = (wave >> 2) * 256;  // O col group

  float m_i = -3.0e38f, l_i = 0.0f;     // persistent softmax state (tid<64)
  FragC acc[16];
#pragma unroll
  for (int t = 0; t < 16; ++t)
#pragma unroll
    for (int r = 0; r < 8; ++r) acc[t].f[r] = 0.0f;

  for (int kt = 0; kt < 16; ++kt) {
    const int cur = kt & 1;
    wait_async();
    __syncthreads();   // tiles for kt resident; buffers for kt+1 free

    if (kt < 15) {     // prefetch next tiles into spare buffers
      tile_to_lds<32, LDK>(K + hb + (size_t)(kt + 1) * 32 * 512,
                           sKb + (cur ^ 1) * 32 * LDK, tid);
      tile_to_lds<32, 512>(V + hb + (size_t)(kt + 1) * 32 * 512,
                           sVn + (cur ^ 1) * 32 * 512, tid);
    }

    // transpose V(cur): sVn[32][512] -> sVt[512][LDV]
    {
      const unsigned short* vn = sVn + cur * 32 * 512;
      for (int e = tid; e < 32 * 256; e += 256) {
        int row = e >> 8, w2 = (e & 255) * 2;
        unsigned d = *(const unsigned*)(vn + row * 512 + w2);
        sVt[w2 * LDV + row] = (unsigned short)(d & 0xFFFFu);
        sVt[(w2 + 1) * LDV + row] = (unsigned short)(d >> 16);
      }
    }

    // S(16x16) = Q(16x512) * K^T  (1/sqrt(W) already folded into Q)
    const unsigned short* sKc = sKb + cur * 32 * LDK;
    FragC s;
#pragma unroll
    for (int r = 0; r < 8; ++r) s.f[r] = 0.0f;
    for (int kk = 0; kk < 16; ++kk) {
      v16bf a = a_frag(sQ, rg, kk * 32, LDQ);
      v16bf bop = b_frag_nk(sKc, cg16, kk * 32, LDK);  // element(w,g)=K[g,w]
      s.v = wmma_bf16(a, bop, s.v);
    }
#pragma unroll
    for (int r = 0; r < 8; ++r)
      sS[(rg + r + 8 * hf) * 32 + cg16 + (lane & 15)] = s.f[r];
    __syncthreads();   // sS + sVt complete

    // online softmax: thread i owns row i
    if (tid < 64) {
      float mn = m_i;
      for (int j = 0; j < 32; ++j) mn = fmaxf(mn, sS[tid * 32 + j]);
      float alpha = __expf(m_i - mn);
      float ps = 0.0f;
      for (int j = 0; j < 32; ++j) {
        float p = __expf(sS[tid * 32 + j] - mn);
        sP[tid * 32 + j] = f32_bf16(p);
        ps += p;
      }
      l_i = l_i * alpha + ps;
      m_i = mn;
      sRow[tid] = alpha;
    }
    __syncthreads();

    // O = O*alpha + P*V   (V^T fragments are contiguous b128 LDS reads)
    float ar[8];
#pragma unroll
    for (int r = 0; r < 8; ++r) ar[r] = sRow[rg + r + 8 * hf];
    v16bf a = a_frag(sP, rg, 0, 32);
#pragma unroll
    for (int t = 0; t < 16; ++t) {
#pragma unroll
      for (int r = 0; r < 8; ++r) acc[t].f[r] *= ar[r];
      v16bf bop = b_frag_nk(sVt, cg256 + t * 16, 0, LDV);  // element(g,w)=V[g,w]
      acc[t].v = wmma_bf16(a, bop, acc[t].v);
    }
  }

  __syncthreads();
  if (tid < 64) sRow[tid] = 1.0f / l_i;
  __syncthreads();
  float ir[8];
#pragma unroll
  for (int r = 0; r < 8; ++r) ir[r] = sRow[rg + r + 8 * hf];
#pragma unroll
  for (int t = 0; t < 16; ++t)
#pragma unroll
    for (int r = 0; r < 8; ++r) {
      int row = q0 + rg + r + 8 * hf;
      int col = cg256 + t * 16 + (lane & 15);
      O[hb + (size_t)row * 512 + col] = f32_bf16(acc[t].f[r] * ir[r]);
    }
}

// -------------------------------------------- out conv + BN + residual (f32)
__global__ void __launch_bounds__(256) out_kernel(
    const unsigned short* __restrict__ A, const float* __restrict__ x,
    const unsigned short* __restrict__ w, const float* __restrict__ bo,
    float* __restrict__ out) {
  __shared__ unsigned short sW[64 * 64];
  __shared__ float sB[64];
  __shared__ unsigned short sX[128 * 72];
  const int tid = threadIdx.x;
  const int bb = blockIdx.y;
  const int s0 = blockIdx.x * 128;

  for (int e = tid; e < 64 * 64; e += 256) sW[e] = w[e];
  if (tid < 64) sB[tid] = bo[tid];
  for (int e = tid; e < 64 * 64; e += 256) {   // dwords: 2 bf16 each
    int c = e >> 6, w2 = (e & 63) * 2;
    unsigned d = *(const unsigned*)(A + (size_t)(bb * 64 + c) * kHW + s0 + w2);
    sX[w2 * 72 + c] = (unsigned short)(d & 0xFFFFu);
    sX[(w2 + 1) * 72 + c] = (unsigned short)(d >> 16);
  }
  __syncthreads();

  const int wave = tid >> 5, lane = tid & 31, hf = lane >> 4;
  const int nloc = wave * 16;
  const int spos = s0 + nloc + (lane & 15);
#pragma unroll
  for (int ot = 0; ot < 4; ++ot) {
    const int ob = ot * 16;
    FragC acc;
#pragma unroll
    for (int r = 0; r < 8; ++r) acc.f[r] = sB[ob + r + 8 * hf];
#pragma unroll
    for (int kk = 0; kk < 2; ++kk) {
      v16bf a = a_frag(sW, ob, kk * 32, 64);
      v16bf bop = b_frag_nk(sX, nloc, kk * 32, 72);
      acc.v = wmma_bf16(a, bop, acc.v);
    }
#pragma unroll
    for (int r = 0; r < 8; ++r) {
      size_t idx = (size_t)(bb * 64 + ob + r + 8 * hf) * kHW + spos;
      out[idx] = acc.f[r] + x[idx];   // residual
    }
  }
}

// ------------------------------------------------------------------ launch
extern "C" void kernel_launch(void* const* d_in, const int* in_sizes, int n_in,
                              void* d_out, int out_size, void* d_ws, size_t ws_size,
                              hipStream_t stream) {
  (void)in_sizes; (void)n_in; (void)out_size; (void)ws_size;
  const float* feature = (const float*)d_in[0];
  const float* wq = (const float*)d_in[1];
  const float* bq = (const float*)d_in[2];
  const float* gq = (const float*)d_in[3];
  const float* eq = (const float*)d_in[4];
  const float* mq = (const float*)d_in[5];
  const float* vq = (const float*)d_in[6];
  const float* wk = (const float*)d_in[7];
  const float* bk = (const float*)d_in[8];
  const float* gk = (const float*)d_in[9];
  const float* ek = (const float*)d_in[10];
  const float* mk = (const float*)d_in[11];
  const float* vk = (const float*)d_in[12];
  const float* wv = (const float*)d_in[13];
  const float* bv = (const float*)d_in[14];
  const float* gv = (const float*)d_in[15];
  const float* ev = (const float*)d_in[16];
  const float* mv = (const float*)d_in[17];
  const float* vv = (const float*)d_in[18];
  const float* wo = (const float*)d_in[19];
  const float* bo = (const float*)d_in[20];
  const float* go = (const float*)d_in[21];
  const float* eo = (const float*)d_in[22];
  const float* mo = (const float*)d_in[23];
  const float* vo = (const float*)d_in[24];

  char* ws = (char*)d_ws;
  unsigned short* fwq = (unsigned short*)(ws + 0);
  unsigned short* fwk = (unsigned short*)(ws + 8192);
  unsigned short* fwv = (unsigned short*)(ws + 16384);
  unsigned short* fwo = (unsigned short*)(ws + 24576);
  float* fbq = (float*)(ws + 32768);
  float* fbk = (float*)(ws + 33024);
  float* fbv = (float*)(ws + 33280);
  float* fbo = (float*)(ws + 33536);
  const size_t matB = (size_t)kHeads * kHW * sizeof(unsigned short);  // 128 MB
  unsigned short* Qb = (unsigned short*)(ws + 65536);
  unsigned short* Kb = (unsigned short*)(ws + 65536 + matB);
  unsigned short* Vb = (unsigned short*)(ws + 65536 + 2 * matB);
  unsigned short* Ab = (unsigned short*)(ws + 65536 + 3 * matB);

  const float qscale = 0.044194173824159216f;  // 1/sqrt(W=512), folded into q
  fold_kernel<<<16, 256, 0, stream>>>(wq, bq, gq, eq, mq, vq, qscale, fwq, fbq);
  fold_kernel<<<16, 256, 0, stream>>>(wk, bk, gk, ek, mk, vk, 1.0f, fwk, fbk);
  fold_kernel<<<16, 256, 0, stream>>>(wv, bv, gv, ev, mv, vv, 1.0f, fwv, fbv);
  fold_kernel<<<16, 256, 0, stream>>>(wo, bo, go, eo, mo, vo, 1.0f, fwo, fbo);

  qkv_kernel<<<dim3(kHW / 128, kB), 256, 0, stream>>>(
      feature, fwq, fwk, fwv, fbq, fbk, fbv, Qb, Kb, Vb);

  size_t shm = (size_t)(64 * LDQ + 2 * 32 * LDK + 2 * 32 * 512 + 512 * LDV +
                        64 * 32) * 2 +
               (size_t)(64 * 32 + 64) * 4;  // 252,160 B
  attn_kernel<<<dim3(kH / 64, kHeads), 256, shm, stream>>>(Qb, Kb, Vb, Ab);

  out_kernel<<<dim3(kHW / 128, kB), 256, 0, stream>>>(Ab, feature, fwo, fbo,
                                                      (float*)d_out);
}
